// LinearNNEncoder_22677427322999
// MI455X (gfx1250) — matrix operand order repaired and checked by reading, not compiled, not run on hardware
//
#include <hip/hip_runtime.h>

typedef __attribute__((ext_vector_type(16))) __bf16 v16bf;
typedef __attribute__((ext_vector_type(8)))  float  v8f;

#define B_    64
#define T_    2048
#define D_    300
#define NEG1  0xBF800000u   // bit pattern of -1.0f

// ws layout (bytes):
//   [0)            Wbf fragments: 19*10*32 frags * 32B       = 194560
//   [194560)       partials: 64*8 slots * 608 floats         = 1245184
//   [1439744)      counts:   64*8 ints                       = 2048
#define WS_WBF_OFF   0
#define WS_PART_OFF  194560
#define WS_CNT_OFF   1439744

union BfVec {
  v16bf v;
  unsigned int u[8];
};

// one-op truncating pack of two f32 -> packed bf16x2 (hot path)
__device__ __forceinline__ unsigned pack2_trunc(unsigned lo, unsigned hi) {
  return __builtin_amdgcn_perm(hi, lo, 0x07060302u);
}

// RNE pack (prep path, off the critical path)
__device__ __forceinline__ unsigned pack2_rne(float lo, float hi) {
  unsigned ulo = __float_as_uint(lo);
  unsigned uhi = __float_as_uint(hi);
  ulo += 0x7FFFu + ((ulo >> 16) & 1u);
  uhi += 0x7FFFu + ((uhi >> 16) & 1u);
  return (uhi & 0xFFFF0000u) | (ulo >> 16);
}

// ---------------- Pass 0: repack W into B-fragment layout (bf16) -------------
// Fragment (j, kk, lane): 16 bf16 = 32B.  lanes 0-15: N=lane,  K=k0..k0+7, k0+16..k0+23
//                                         lanes 16-31: N=lane-16, K=k0+8..k0+15, k0+24..k0+31
__global__ __launch_bounds__(256) void prep_w_kernel(
    const float* __restrict__ W, unsigned* __restrict__ Wbf) {
  int t = blockIdx.x * 256 + threadIdx.x;
  if (t >= 19 * 10 * 32) return;
  int lane = t & 31;
  int kk   = (t >> 5) % 10;
  int j    = t / 320;
  int nn   = lane & 15;
  int hi   = (lane >> 4) & 1;
  int row  = j * 16 + nn;
  int kb   = kk * 32 + hi * 8;
  unsigned u[8];
#pragma unroll
  for (int i = 0; i < 8; ++i) {
    float f0 = 0.f, f1 = 0.f;
    int k0 = kb + 2 * i + (i >= 4 ? 8 : 0);
    if (row < D_) {
      if (k0     < D_) f0 = W[row * D_ + k0];
      if (k0 + 1 < D_) f1 = W[row * D_ + k0 + 1];
    }
    u[i] = pack2_rne(f0, f1);
  }
  unsigned* dst = Wbf + (size_t)t * 8;
  *(uint4*)dst       = make_uint4(u[0], u[1], u[2], u[3]);
  *((uint4*)dst + 1) = make_uint4(u[4], u[5], u[6], u[7]);
}

// ---------------- Pass 1: GEMM + GELU + masked sum/sumsq ---------------------
// grid (8 T-chunks, 64 batches), 608 threads = 19 waves; wave w owns N-tile j=w.
__global__ __launch_bounds__(608) void encoder_main_kernel(
    const float* __restrict__ x, const unsigned* __restrict__ Wbf,
    const float* __restrict__ bias, float* __restrict__ part,
    int* __restrict__ cntws) {
  __shared__ __align__(16) unsigned Apack[10][32][8];  // A fragments (bf16 pairs)
  __shared__ unsigned rowNE[16];                       // per-tile row "any != -1" bits

  const int tc   = blockIdx.x;        // token chunk (256 tokens)
  const int bb   = blockIdx.y;        // batch
  const int tid  = threadIdx.x;
  const int w    = tid >> 5;          // wave = N tile j (0..18)
  const int lane = tid & 31;
  const int nn   = lane & 15;
  const bool hi  = lane >= 16;

  // B fragments for this wave's N tile: loaded once, reused for all 16 M tiles
  v16bf bw[10];
#pragma unroll
  for (int kk = 0; kk < 10; ++kk) {
    const uint4* src = (const uint4*)(Wbf + ((size_t)(w * 10 + kk) * 32 + lane) * 8);
    uint4 p0 = src[0], p1 = src[1];
    BfVec t;
    t.u[0] = p0.x; t.u[1] = p0.y; t.u[2] = p0.z; t.u[3] = p0.w;
    t.u[4] = p1.x; t.u[5] = p1.y; t.u[6] = p1.z; t.u[7] = p1.w;
    bw[kk] = t.v;
  }
  const int   col     = w * 16 + nn;
  const float colBias = (col < D_) ? bias[col] : 0.0f;

  // one-time LDS init: per-tile row masks + permanent zero K-tail of kk=9 frags
  if (tid < 16)  rowNE[tid] = 0u;
  if (tid < 128) Apack[9][tid >> 2][4 + (tid & 3)] = 0u;
  __syncthreads();

  // staging role: 608 threads = 16 rows x 38 chunks of 8 floats (300 -> 37.5)
  const int  sr    = tid / 38;            // row in tile
  const int  ck    = tid % 38;            // k chunk
  const int  k8    = ck * 8;
  const int  skk   = k8 >> 5;             // fragment K index
  const int  srem  = k8 & 31;
  const int  shi   = (srem == 8 || srem == 24) ? 1 : 0;
  const int  sui   = (srem < 16) ? 0 : 4; // uint index inside fragment
  const int  slane = shi * 16 + sr;
  const bool slast = (ck == 37);          // covers k 296..299 (+4 pad)

  const float* xb = x + ((size_t)bb * T_ + (size_t)tc * 256) * D_;

  float s = 0.0f, q = 0.0f;
  int   cnt = 0;

  for (int i = 0; i < 16; ++i) {
    // ---- stage x tile i: f32 -> bf16 A fragments in LDS + validity bits ----
    {
      const unsigned* xr = (const unsigned*)(xb + (size_t)(i * 16 + sr) * D_);
      if (i < 15)
        __builtin_prefetch((const float*)xr + 16 * D_ + k8, 0, 1);
      uint4 c0 = *(const uint4*)(xr + k8);
      uint4 c1 = make_uint4(0u, 0u, 0u, 0u);
      unsigned ne = (c0.x ^ NEG1) | (c0.y ^ NEG1) | (c0.z ^ NEG1) | (c0.w ^ NEG1);
      if (!slast) {
        c1 = *(const uint4*)(xr + k8 + 4);
        ne |= (c1.x ^ NEG1) | (c1.y ^ NEG1) | (c1.z ^ NEG1) | (c1.w ^ NEG1);
      }
      if (ne) atomicOr(&rowNE[i], 1u << sr);   // bitwise OR: deterministic
      uint4 pk;
      pk.x = pack2_trunc(c0.x, c0.y);
      pk.y = pack2_trunc(c0.z, c0.w);
      pk.z = pack2_trunc(c1.x, c1.y);
      pk.w = pack2_trunc(c1.z, c1.w);
      *(uint4*)&Apack[skk][slane][sui] = pk;
    }
    __syncthreads();

    // ---- compute: 10 x wmma_bf16, B pinned in registers ----
    unsigned vm = rowNE[i] & 0xFFFFu;    // bit r set -> row r is VALID
    if (w == 0 && lane == 0) cnt += __popc(vm);
    v8f c = {};
#pragma unroll
    for (int kk = 0; kk < 10; ++kk) {
      const uint4* ap = (const uint4*)&Apack[kk][lane][0];
      uint4 p0 = ap[0], p1 = ap[1];
      BfVec av;
      av.u[0] = p0.x; av.u[1] = p0.y; av.u[2] = p0.z; av.u[3] = p0.w;
      av.u[4] = p1.x; av.u[5] = p1.y; av.u[6] = p1.z; av.u[7] = p1.w;
      c = __builtin_amdgcn_wmma_f32_16x16x32_bf16(
          false, av.v, false, bw[kk], (short)0, c, false, false);
    }

    // ---- epilogue: bias + GELU (exp/rcp TRANS form) + masked accumulation ----
#pragma unroll
    for (int rr = 0; rr < 8; ++rr) {
      int   row = rr + (hi ? 8 : 0);     // C layout rows
      float m   = ((vm >> row) & 1u) ? 1.0f : 0.0f;
      float v   = c[rr] + colBias;
      float z   = v * fmaf(0.0356774081f, v * v, 0.7978845608f);
      float e   = __expf(2.0f * z);
      float r   = __builtin_amdgcn_rcpf(1.0f + e);
      float y   = (v - v * r) * m;       // v*(1-r) == 0.5*v*(1+tanh(z))
      s += y;
      q  = fmaf(y, y, q);
    }
    __syncthreads();   // Apack reuse guard
  }

  // merge lane pairs (L, L+16) -> full column partials, write per-block slot
  s += __shfl_xor(s, 16, 32);
  q += __shfl_xor(q, 16, 32);
  const int slot = bb * 8 + tc;
  if (lane < 16) {
    float* p = part + (size_t)slot * 608;
    p[w * 16 + nn]       = s;
    p[304 + w * 16 + nn] = q;
  }
  if (w == 0 && lane == 0) cntws[slot] = cnt;
}

// ---------------- Pass 2: deterministic reduction -> std/mean ----------------
__global__ __launch_bounds__(320) void finalize_kernel(
    const float* __restrict__ part, const int* __restrict__ cntws,
    float* __restrict__ out) {
  int bb  = blockIdx.x;
  int col = threadIdx.x;
  if (col >= D_) return;
  float S = 0.f, Q = 0.f;
  int   n = 0;
#pragma unroll
  for (int tc = 0; tc < 8; ++tc) {
    int slot = bb * 8 + tc;
    S += part[(size_t)slot * 608 + col];
    Q += part[(size_t)slot * 608 + 304 + col];
    n += cntws[slot];
  }
  float fn   = (float)n;
  float mean = S / fn;
  float var  = (Q - S * mean) / fmaxf(fn - 1.0f, 1.0f);
  float stdv = (n > 1) ? sqrtf(fmaxf(var, 0.0f)) : 0.0f;
  if (__builtin_isnan(mean)) mean = 0.0f;
  if (__builtin_isnan(stdv)) stdv = 0.0f;
  float* ob = out + (size_t)bb * (2 * D_);
  ob[col]      = stdv;
  ob[D_ + col] = mean;
}

extern "C" void kernel_launch(void* const* d_in, const int* in_sizes, int n_in,
                              void* d_out, int out_size, void* d_ws, size_t ws_size,
                              hipStream_t stream) {
  (void)in_sizes; (void)n_in; (void)out_size; (void)ws_size;
  const float* x    = (const float*)d_in[0];  // [64, 2048, 300]
  const float* W    = (const float*)d_in[1];  // [300, 300]
  const float* bias = (const float*)d_in[2];  // [300]
  float* out = (float*)d_out;                 // [64, 600]

  unsigned* Wbf  = (unsigned*)((char*)d_ws + WS_WBF_OFF);
  float*    part = (float*)   ((char*)d_ws + WS_PART_OFF);
  int*      cnt  = (int*)     ((char*)d_ws + WS_CNT_OFF);

  prep_w_kernel<<<24, 256, 0, stream>>>(W, Wbf);
  encoder_main_kernel<<<dim3(8, B_), 608, 0, stream>>>(x, Wbf, bias, part, cnt);
  finalize_kernel<<<B_, 320, 0, stream>>>(part, cnt, out);
}
